// FocalLoss_57853209477363
// MI455X (gfx1250) — compile-verified, matrix operand order checked by reading, not run on hardware
//
#include <hip/hip_runtime.h>
#include <hip/hip_bf16.h>
#include <math.h>

typedef __attribute__((ext_vector_type(16))) _Float16 v16h;
typedef __attribute__((ext_vector_type(8)))  _Float16 v8h;
typedef __attribute__((ext_vector_type(4)))  _Float16 v4h;
typedef __attribute__((ext_vector_type(8)))  float    v8f;

// Problem constants (match reference)
constexpr int   Bsz       = 4096;
constexpr int   Csz       = 5000;
constexpr float GAMMA_POS = 1.05f;
constexpr float GAMMA_NG  = 1.0f;
constexpr float WEIGHT    = 1.0f;
constexpr float CLIP      = 0.05f;
constexpr float EPSF      = 1e-8f;

// Padded f16 operand layouts in workspace (zero-padded K and N edges)
constexpr int KP = 5024;              // K stride in halves (5000 -> mult. of 32)
constexpr int NP = 5120;              // coT rows (5000 -> mult. of 128)

// GEMM tiling
constexpr int BM  = 128;              // block tile M
constexpr int BN  = 128;              // block tile N
constexpr int BK  = 32;               // k-step (WMMA K)
constexpr int LDH = 40;               // LDS row stride in halves (80B = 20 banks)

// Workspace layout (bytes)
constexpr size_t YH_OFF  = 0;
constexpr size_t YH_SZ   = (size_t)Bsz * KP * 2;          // 41,156,608
constexpr size_t COT_OFF = YH_OFF + YH_SZ;
constexpr size_t COT_SZ  = (size_t)NP * KP * 2;           // 51,445,760
constexpr size_t S_OFF   = COT_OFF + COT_SZ;

// ---------------- CDNA5 async global->LDS copy helpers ----------------
__device__ __forceinline__ void async_copy_b128(uint32_t lds_byte_off,
                                                const void* gptr) {
  unsigned long long ga = (unsigned long long)(uintptr_t)gptr;
  asm volatile("global_load_async_to_lds_b128 %0, %1, off"
               :: "v"(lds_byte_off), "v"(ga)
               : "memory");
}
__device__ __forceinline__ void wait_asynccnt0() {
  asm volatile("s_wait_asynccnt 0x0" ::: "memory");
}
__device__ __forceinline__ uint32_t lds_off(const void* p) {
  return (uint32_t)(uintptr_t)p;   // low 32 bits of flat addr = LDS byte address
}

__device__ inline float blockReduceSum(float v, float* sh) {
  #pragma unroll
  for (int off = 16; off > 0; off >>= 1) v += __shfl_down(v, off, 32);
  const int lane = threadIdx.x & 31;
  const int wave = threadIdx.x >> 5;
  if (lane == 0) sh[wave] = v;
  __syncthreads();
  const int nw = blockDim.x >> 5;
  v = (threadIdx.x < (unsigned)nw) ? sh[threadIdx.x] : 0.0f;
  if (wave == 0) {
    #pragma unroll
    for (int off = 16; off > 0; off >>= 1) v += __shfl_down(v, off, 32);
  }
  return v; // valid in thread 0
}

// ---- y (fp32, {0,1}) -> yh (f16, [Bsz][KP], K zero-padded) ----
__global__ void convert_y_kernel(const float* __restrict__ y,
                                 _Float16* __restrict__ yh) {
  const int m = blockIdx.x;
  for (int s4 = threadIdx.x; s4 < KP / 4; s4 += blockDim.x) {
    const int kG = s4 * 4;
    float4 v = make_float4(0.f, 0.f, 0.f, 0.f);
    if (kG < Csz)   // Csz%4==0 -> float4 fully valid or fully pad
      v = *(const float4*)&y[(size_t)m * Csz + kG];
    v4h h;
    h[0] = (_Float16)v.x; h[1] = (_Float16)v.y;
    h[2] = (_Float16)v.z; h[3] = (_Float16)v.w;
    *(v4h*)&yh[(size_t)m * KP + kG] = h;
  }
}

// ---- co (fp32, [C][C] k-major) -> coT (f16, [NP][KP] n-major, zero-padded) ----
__global__ void transpose_co_kernel(const float* __restrict__ co,
                                    _Float16* __restrict__ coT) {
  __shared__ float t[64][65];
  const int tid = threadIdx.x;
  const int j0 = blockIdx.x * 64;       // output-row (class column) tile
  const int k0 = blockIdx.y * 64;       // k tile
  #pragma unroll 4
  for (int p = 0; p < 16; ++p) {
    const int ky = (tid >> 6) + p * 4;
    const int jx = tid & 63;
    const int k = k0 + ky, j = j0 + jx;
    t[ky][jx] = (k < Csz && j < Csz) ? co[(size_t)k * Csz + j] : 0.0f;
  }
  __syncthreads();
  #pragma unroll 4
  for (int p = 0; p < 16; ++p) {
    const int jy = (tid >> 6) + p * 4;
    const int kx = tid & 63;
    const int j = j0 + jy, k = k0 + kx;
    if (k < KP)                          // j < NP by grid construction
      coT[(size_t)j * KP + k] = (_Float16)t[kx][jy];
  }
}

// ---- s[k] = sum_j co[k, j]; also zero the output accumulator ----
__global__ void co_rowsum_kernel(const float* __restrict__ co,
                                 float* __restrict__ s,
                                 float* __restrict__ out) {
  __shared__ float sh[8];
  const int k = blockIdx.x;
  float v = 0.0f;
  for (int j = threadIdx.x; j < Csz; j += blockDim.x)
    v += co[(size_t)k * Csz + j];
  v = blockReduceSum(v, sh);
  if (threadIdx.x == 0) {
    s[k] = v;
    if (k == 0) out[0] = 0.0f;
  }
}

// ---- invDenom[b] = 1 / sum_k y[b,k]*s[k]  (row-norm factors cancel) ----
__global__ void denom_kernel(const float* __restrict__ y,
                             const float* __restrict__ s,
                             float* __restrict__ invDenom) {
  __shared__ float sh[8];
  const int b = blockIdx.x;
  float v = 0.0f;
  for (int k = threadIdx.x; k < Csz; k += blockDim.x)
    v += y[(size_t)b * Csz + k] * s[k];
  v = blockReduceSum(v, sh);
  if (threadIdx.x == 0) invDenom[b] = 1.0f / v;
}

// ---- fused (y @ co)/denom -> focal loss -> scalar, WMMA + async double-buffer ----
__global__ __launch_bounds__(256, 1)
void focal_wmma_kernel(const float* __restrict__ x,
                       const float* __restrict__ y,
                       const _Float16* __restrict__ yh,
                       const _Float16* __restrict__ coT,
                       const float* __restrict__ invDenom,
                       float* __restrict__ out) {
  __shared__ _Float16 ldsA[2][BM][LDH];   // y tile (m-major, k inner)
  __shared__ _Float16 ldsB[2][BN][LDH];   // coT tile (n-major, k inner)
  __shared__ float    shRed[8];

  const int tid  = threadIdx.x;
  const int lane = tid & 31;
  const int wave = tid >> 5;              // 8 waves: 4 along M x 2 along N
  const int wm   = wave >> 1;             // 0..3  -> 32 rows each
  const int wn   = wave & 1;              // 0..1  -> 64 cols each
  const int mBase = blockIdx.y * BM;
  const int nBase = blockIdx.x * BN;

  const int r15 = lane & 15;
  const int ko  = (lane >> 4) * 8;        // K-chunk select per ISA 16-bit layout

  // Per-thread async-copy assignments: 512 b128 chunks per operand tile.
  // chunk c: row = c>>2 (0..127), ch = c&3 (16B chunk within 64B of k-data)
  const int cA0 = tid, cA1 = tid + 256;

  v8f acc[2][4] = {};

  auto issue_tile = [&](int kt, int buf) {
    const int kBase = kt * BK;
    {
      const int row = cA0 >> 2, ch = cA0 & 3;
      async_copy_b128(lds_off(&ldsA[buf][row][ch * 8]),
                      &yh[(size_t)(mBase + row) * KP + kBase + ch * 8]);
      async_copy_b128(lds_off(&ldsB[buf][row][ch * 8]),
                      &coT[(size_t)(nBase + row) * KP + kBase + ch * 8]);
    }
    {
      const int row = cA1 >> 2, ch = cA1 & 3;
      async_copy_b128(lds_off(&ldsA[buf][row][ch * 8]),
                      &yh[(size_t)(mBase + row) * KP + kBase + ch * 8]);
      async_copy_b128(lds_off(&ldsB[buf][row][ch * 8]),
                      &coT[(size_t)(nBase + row) * KP + kBase + ch * 8]);
    }
  };

  const int KT = (Csz + BK - 1) / BK;     // 157
  issue_tile(0, 0);

  for (int kt = 0; kt < KT; ++kt) {
    const int cur = kt & 1;
    wait_asynccnt0();                     // this wave's copies into `cur` done
    __syncthreads();                      // all waves' copies visible; prior reads drained
    if (kt + 1 < KT) issue_tile(kt + 1, cur ^ 1);

    // ---- fragments per ISA 16-bit A/B VGPR layout (all 16B-aligned b128) ----
    v16h afr[2], bfr[4];
    #pragma unroll
    for (int t = 0; t < 2; ++t) {
      const int row = wm * 32 + t * 16 + r15;
      v8h lo = *(const v8h*)&ldsA[cur][row][ko];
      v8h hi = *(const v8h*)&ldsA[cur][row][ko + 16];
      #pragma unroll
      for (int i = 0; i < 8; ++i) { afr[t][i] = lo[i]; afr[t][8 + i] = hi[i]; }
    }
    #pragma unroll
    for (int t = 0; t < 4; ++t) {
      const int col = wn * 64 + t * 16 + r15;
      v8h lo = *(const v8h*)&ldsB[cur][col][ko];
      v8h hi = *(const v8h*)&ldsB[cur][col][ko + 16];
      #pragma unroll
      for (int i = 0; i < 8; ++i) { bfr[t][i] = lo[i]; bfr[t][8 + i] = hi[i]; }
    }

    // ---- 8x v_wmma_f32_16x16x32_f16 per wave per k-step ----
    #pragma unroll
    for (int tm = 0; tm < 2; ++tm)
      #pragma unroll
      for (int tn = 0; tn < 4; ++tn)
        acc[tm][tn] = __builtin_amdgcn_wmma_f32_16x16x32_f16(
            false, afr[tm], false, bfr[tn],
            (short)0, acc[tm][tn], false, false);
  }

  // ---- fused focal-loss epilogue ----
  // C/D layout: lanes 0-15 -> M=v, lanes 16-31 -> M=v+8; N = lane&15.
  float psum = 0.0f;
  #pragma unroll
  for (int tm = 0; tm < 2; ++tm) {
    #pragma unroll
    for (int tn = 0; tn < 4; ++tn) {
      #pragma unroll
      for (int v = 0; v < 8; ++v) {
        const int bRow = mBase + wm * 32 + tm * 16 + v + ((lane >> 4) << 3);
        const int jCol = nBase + wn * 64 + tn * 16 + r15;
        if (jCol < Csz) {
          const float attn = acc[tm][tn][v] * invDenom[bRow];
          const size_t idx = (size_t)bRow * Csz + jCol;
          const float xv = x[idx];
          const float yv = y[idx];
          const float xs = 1.0f / (1.0f + __expf(-xv));
          const float xs_pos = xs * (GAMMA_POS - attn);
          float xs_neg = 1.0f - xs;
          xs_neg = fminf(xs_neg + CLIP, 1.0f);
          xs_neg = fminf((attn == 0.0f) ? xs_neg : xs_neg * GAMMA_NG, 1.0f);
          const float los_pos = yv * __logf(fmaxf(xs_pos, EPSF));
          const float los_neg = (1.0f - yv) * __logf(fmaxf(xs_neg, EPSF));
          const float loss = (los_pos + los_neg) * WEIGHT;
          const float pt = xs_pos * yv + xs_neg * (1.0f - yv);
          const float osw = 1.0f - pt;   // pow(1-pt, WEIGHT), WEIGHT==1
          psum += loss * osw;
        }
      }
    }
  }

  const float total = blockReduceSum(psum, shRed);
  if (tid == 0) atomicAdd(out, -total);
}

extern "C" void kernel_launch(void* const* d_in, const int* in_sizes, int n_in,
                              void* d_out, int out_size, void* d_ws, size_t ws_size,
                              hipStream_t stream) {
  const float* x  = (const float*)d_in[0];   // [B, C] fp32
  const float* y  = (const float*)d_in[1];   // [B, C] fp32 (binary)
  const float* co = (const float*)d_in[2];   // [C, C] fp32
  // d_in[3] = epoch (int scalar), unused by the math

  float*    out      = (float*)d_out;
  char*     ws       = (char*)d_ws;
  _Float16* yh       = (_Float16*)(ws + YH_OFF);
  _Float16* coT      = (_Float16*)(ws + COT_OFF);
  float*    s        = (float*)(ws + S_OFF);       // Csz sums (padded to 5120)
  float*    invDenom = s + 5120;                   // Bsz inverse denominators

  convert_y_kernel  <<<Bsz, 256, 0, stream>>>(y, yh);
  transpose_co_kernel<<<dim3(NP / 64, (KP + 63) / 64), 256, 0, stream>>>(co, coT);
  co_rowsum_kernel  <<<Csz, 256, 0, stream>>>(co, s, out);
  denom_kernel      <<<Bsz, 256, 0, stream>>>(y, s, invDenom);

  dim3 grid(NP / BN, Bsz / BM);                    // 40 x 32 blocks
  focal_wmma_kernel<<<grid, 256, 0, stream>>>(x, y, yh, coT, invDenom, out);
}
